// ResidualVectorQuantizer_29892972380619
// MI455X (gfx1250) — matrix operand order repaired and checked by reading, not compiled, not run on hardware
//
#include <hip/hip_runtime.h>
#include <hip/hip_bf16.h>
#include <stdint.h>

// ---------------------------------------------------------------------------
// Residual Vector Quantizer, 3 levels.  B=32768, D=256, K=8192.
// Distance scan as bf16 WMMA GEMM + fused argmax. Codebook tiles staged in
// LDS via CDNA5 async-to-LDS copies (ASYNCcnt), double buffered, so WMMAs
// (DScnt-tracked LDS reads) overlap the next stage's global->LDS traffic.
// ---------------------------------------------------------------------------

#define RVQ_B 32768
#define RVQ_D 256
#define RVQ_K 8192
#define RVQ_L 3

typedef __attribute__((ext_vector_type(16))) __bf16 v16bf;
typedef __attribute__((ext_vector_type(8)))  __bf16 v8bf;
typedef __attribute__((ext_vector_type(8)))  float  v8f;

// ---- CDNA5 async global->LDS copy (tracked by ASYNCcnt, not LOAD/DScnt) ----
__device__ __forceinline__ void async_ld_b128(uint32_t lds_byte_addr, uint64_t gaddr) {
    asm volatile("global_load_async_to_lds_b128 %0, %1, off"
                 :: "v"(lds_byte_addr), "v"(gaddr) : "memory");
}
__device__ __forceinline__ void wait_asynccnt0() {
    asm volatile("s_wait_asynccnt 0x0" ::: "memory");
}

// ---------------------------------------------------------------------------
// Block-wide sum reduction (blockDim.x == 256, wave32).  Result valid on tid 0.
// ---------------------------------------------------------------------------
__device__ __forceinline__ float block_reduce_sum_256(float v) {
    __shared__ float s_red[8];
    const int lane = threadIdx.x & 31;
    const int w    = threadIdx.x >> 5;
    #pragma unroll
    for (int o = 16; o > 0; o >>= 1) v += __shfl_down(v, o, 32);
    if (lane == 0) s_red[w] = v;
    __syncthreads();
    if (w == 0) {
        v = (lane < 8) ? s_red[lane] : 0.0f;
        #pragma unroll
        for (int o = 4; o > 0; o >>= 1) v += __shfl_down(v, o, 32);
    }
    return v;
}

// ---------------------------------------------------------------------------
// Convert one codeword (per block) of one codebook to bf16 and emit 0.5*||c||^2.
// grid = 3*K blocks, block = 256 threads.
// ---------------------------------------------------------------------------
__global__ void rvq_convert_kernel(const float* __restrict__ cb0,
                                   const float* __restrict__ cb1,
                                   const float* __restrict__ cb2,
                                   __bf16* __restrict__ cbbf,     // [3][K][D]
                                   float*  __restrict__ cnorm) {  // [3][K]
    const int lvl = blockIdx.x >> 13;          // / 8192
    const int k   = blockIdx.x & (RVQ_K - 1);
    const int d   = threadIdx.x;
    const float* src = (lvl == 0) ? cb0 : (lvl == 1) ? cb1 : cb2;
    const float v = src[(size_t)k * RVQ_D + d];
    cbbf[((size_t)lvl * RVQ_K + k) * RVQ_D + d] = (__bf16)v;
    float ssq = block_reduce_sum_256(v * v);
    if (threadIdx.x == 0) cnorm[lvl * RVQ_K + k] = 0.5f * ssq;
}

// ---------------------------------------------------------------------------
// Fused GEMM + argmax kernel for one level.
//   score[b][k] = r_b . c_k - 0.5*||c_k||^2      (argmax == argmin distance)
// grid = B/128 = 256 blocks, block = 256 threads = 8 waves.
// Each wave owns 16 rows (A fragments in registers).  Codebook streamed in
// 64-column stages through a double-buffered 2x32KB LDS tile shared by all
// 8 waves; each stage = 4 column tiles x 8 WMMA (16x16x32 bf16) per wave.
// ---------------------------------------------------------------------------
__global__ void __launch_bounds__(256)
rvq_argmax_kernel(const float*  __restrict__ resid,   // [B][D] fp32
                  const __bf16* __restrict__ cbbf,    // [K][D] bf16 (this level)
                  const float*  __restrict__ cnorm,   // [K] 0.5*||c||^2
                  int*          __restrict__ idx_out) // [B]
{
    // 64KB region: residual tile during prologue, then codebook double buffer.
    __shared__ __attribute__((aligned(64))) char smem[64 * 1024 + 512];
    __bf16* s_r  = (__bf16*)smem;               // [128][256] (prologue only)
    __bf16* s_cb = (__bf16*)smem;               // [2][64][256] (main loop)
    float*  s_cn = (float*)(smem + 64 * 1024);  // [2][64] biases

    const int tid     = threadIdx.x;
    const int rowBase = blockIdx.x * 128;

    // --- Prologue: stage 128x256 residual tile into LDS as bf16.
    #pragma unroll 4
    for (int i = tid; i < 128 * RVQ_D; i += 256) {
        const int r = i >> 8;
        const int d = i & (RVQ_D - 1);
        s_r[i] = (__bf16)resid[(size_t)(rowBase + r) * RVQ_D + d];
    }
    __syncthreads();

    const int lane = tid & 31;
    const int wv   = tid >> 5;
    const int m    = lane & 15;   // A-row / B-col residue
    const int half = lane >> 4;   // lane-group select
    const int rowLocal = wv * 16 + m;

    // A fragments (contiguous k order, matching B):
    //   lanes 0-15:  K 0..15
    //   lanes 16-31: K 16..31
    v16bf afrag[8];
    #pragma unroll
    for (int c = 0; c < 8; ++c) {
        const __bf16* pr = &s_r[rowLocal * RVQ_D + c * 32];
        union { v16bf v; v8bf h[2]; } u;
        u.h[0] = *(const v8bf*)(pr + half * 16);
        u.h[1] = *(const v8bf*)(pr + half * 16 + 8);
        afrag[c] = u.v;
    }
    __syncthreads();   // all waves done reading s_r before s_cb overwrites it

    float best[8];
    int   bidx[8];
    #pragma unroll
    for (int j = 0; j < 8; ++j) { best[j] = -3.4e38f; bidx[j] = 0; }

    const uint32_t ldsCbBase = (uint32_t)(uintptr_t)s_cb;  // LDS byte addr
    const int S = RVQ_K / 64;                              // 128 stages

    // Stage issue: 64 codewords = contiguous 32KB; 256 threads x 8 x b128.
    auto issue_stage = [&](int s, int buf) {
        const uint64_t gsrc = (uint64_t)(uintptr_t)cbbf + (uint64_t)s * 32768u;
        const uint32_t lbase = ldsCbBase + (uint32_t)buf * 32768u;
        #pragma unroll
        for (int p = 0; p < 8; ++p) {
            const uint32_t o = (uint32_t)(p * 4096 + tid * 16);
            async_ld_b128(lbase + o, gsrc + o);
        }
        if (tid < 64) s_cn[buf * 64 + tid] = cnorm[s * 64 + tid];
    };

    issue_stage(0, 0);
    wait_asynccnt0();
    __syncthreads();

    int cur = 0;
    for (int s = 0; s < S; ++s) {
        if (s + 1 < S) issue_stage(s + 1, cur ^ 1);

        // --- Compute: 4 column tiles of 16 codewords from LDS buffer `cur`.
        const __bf16* cbbuf = s_cb + cur * (64 * RVQ_D);
        #pragma unroll
        for (int tl = 0; tl < 4; ++tl) {
            const int colLocal = tl * 16 + m;
            // B fragment (32x16 bf16): lane holds column colLocal, 16
            // contiguous d values starting at half*16 within each 32-chunk.
            const __bf16* bp = cbbuf + colLocal * RVQ_D + half * 16;

            v8f cfrag = {};
            #pragma unroll
            for (int c = 0; c < 8; ++c) {
                v16bf bfrag = *(const v16bf*)(bp + (c << 5));
                cfrag = __builtin_amdgcn_wmma_f32_16x16x32_bf16(
                    false, afrag[c], false, bfrag, (short)0, cfrag, false, false);
            }

            const float bias = s_cn[cur * 64 + colLocal];
            const int   col  = (s << 6) + colLocal;
            // C layout: vgpr j, lanes 0-15 -> row j; lanes 16-31 -> row j+8.
            #pragma unroll
            for (int j = 0; j < 8; ++j) {
                const float sc = cfrag[j] - bias;
                if (sc > best[j]) { best[j] = sc; bidx[j] = col; }
            }
        }

        if (s + 1 < S) wait_asynccnt0();
        __syncthreads();
        cur ^= 1;
    }

    // --- Reduce over the 16 lanes holding each row (tie: lower index wins,
    // matching argmin-first semantics).
    #pragma unroll
    for (int off = 1; off <= 8; off <<= 1) {
        #pragma unroll
        for (int j = 0; j < 8; ++j) {
            const float ob = __shfl_xor(best[j], off, 32);
            const int   oi = __shfl_xor(bidx[j], off, 32);
            if (ob > best[j] || (ob == best[j] && oi < bidx[j])) {
                best[j] = ob; bidx[j] = oi;
            }
        }
    }
    if (m == 0) {
        const int rb = rowBase + wv * 16 + half * 8;
        #pragma unroll
        for (int j = 0; j < 8; ++j) idx_out[rb + j] = bidx[j];
    }
}

// ---------------------------------------------------------------------------
// Gather q = cb[idx], r -= q, emit per-block sum of new r^2.
// grid = B blocks, block = 256 threads (one per d).
// ---------------------------------------------------------------------------
__global__ void rvq_update_kernel(const float* __restrict__ cb,     // [K][D] fp32
                                  const int*   __restrict__ idx,    // [B]
                                  float*       __restrict__ resid,  // [B][D]
                                  float*       __restrict__ partial)// [B]
{
    const int b = blockIdx.x;
    const int d = threadIdx.x;
    const int k = idx[b];
    const float q = cb[(size_t)k * RVQ_D + d];
    const size_t off = (size_t)b * RVQ_D + d;
    const float r = resid[off] - q;
    resid[off] = r;
    float s = block_reduce_sum_256(r * r);
    if (threadIdx.x == 0) partial[b] = s;
}

// Deterministic fixed-order reduction of B partials -> acc[level].
__global__ void rvq_reduce_kernel(const float* __restrict__ partial, int n,
                                  float* __restrict__ out) {
    float v = 0.0f;
    for (int i = threadIdx.x; i < n; i += 256) v += partial[i];
    v = block_reduce_sum_256(v);
    if (threadIdx.x == 0) *out = v;
}

// quantized_sum = x - final_residual
__global__ void rvq_qsum_kernel(const float* __restrict__ x,
                                const float* __restrict__ resid,
                                float* __restrict__ out) {
    const size_t i = (size_t)blockIdx.x * 256 + threadIdx.x;
    out[i] = x[i] - resid[i];
}

// indices (int) -> float, concatenated after quantized_sum.
__global__ void rvq_idxf_kernel(const int* __restrict__ idx,
                                float* __restrict__ out) {
    const int i = blockIdx.x * 256 + threadIdx.x;
    out[i] = (float)idx[i];
}

// scalars: recon = mean(r3^2); commit = sum_l mean(r_{l+1}^2); total.
__global__ void rvq_scalars_kernel(const float* __restrict__ acc,
                                   float* __restrict__ out) {
    const float inv = 1.0f / (float)((size_t)RVQ_B * RVQ_D);
    const float recon  = acc[2] * inv;
    const float commit = (acc[0] + acc[1] + acc[2]) * inv;
    out[0] = recon;
    out[1] = commit;
    out[2] = recon + 0.25f * commit;
}

// ---------------------------------------------------------------------------
extern "C" void kernel_launch(void* const* d_in, const int* in_sizes, int n_in,
                              void* d_out, int out_size, void* d_ws, size_t ws_size,
                              hipStream_t stream) {
    const float* x = (const float*)d_in[0];
    const float* cb_f32[RVQ_L] = { (const float*)d_in[1],
                                   (const float*)d_in[2],
                                   (const float*)d_in[3] };

    // Workspace layout (bytes):
    char* ws = (char*)d_ws;
    float*  resid   = (float*)ws;                                   // 32 MB
    size_t  off     = (size_t)RVQ_B * RVQ_D * sizeof(float);
    __bf16* cbbf    = (__bf16*)(ws + off);                          // 12 MB
    off += (size_t)RVQ_L * RVQ_K * RVQ_D * sizeof(__bf16);
    float*  cnorm   = (float*)(ws + off);                           // 96 KB
    off += (size_t)RVQ_L * RVQ_K * sizeof(float);
    int*    idx     = (int*)(ws + off);                             // 384 KB
    off += (size_t)RVQ_L * RVQ_B * sizeof(int);
    float*  partial = (float*)(ws + off);                           // 128 KB
    off += (size_t)RVQ_B * sizeof(float);
    float*  acc     = (float*)(ws + off);                           // 3 floats

    // residual <- x
    hipMemcpyAsync(resid, x, (size_t)RVQ_B * RVQ_D * sizeof(float),
                   hipMemcpyDeviceToDevice, stream);

    // Codebooks -> bf16 + 0.5*||c||^2
    rvq_convert_kernel<<<RVQ_L * RVQ_K, 256, 0, stream>>>(
        cb_f32[0], cb_f32[1], cb_f32[2], cbbf, cnorm);

    for (int l = 0; l < RVQ_L; ++l) {
        rvq_argmax_kernel<<<RVQ_B / 128, 256, 0, stream>>>(
            resid,
            cbbf + (size_t)l * RVQ_K * RVQ_D,
            cnorm + (size_t)l * RVQ_K,
            idx + (size_t)l * RVQ_B);
        rvq_update_kernel<<<RVQ_B, 256, 0, stream>>>(
            cb_f32[l], idx + (size_t)l * RVQ_B, resid, partial);
        rvq_reduce_kernel<<<1, 256, 0, stream>>>(partial, RVQ_B, acc + l);
    }

    float* out = (float*)d_out;
    rvq_qsum_kernel<<<RVQ_B, 256, 0, stream>>>(x, resid, out);
    rvq_idxf_kernel<<<(RVQ_L * RVQ_B) / 256, 256, 0, stream>>>(
        idx, out + (size_t)RVQ_B * RVQ_D);
    rvq_scalars_kernel<<<1, 1, 0, stream>>>(
        acc, out + (size_t)RVQ_B * RVQ_D + (size_t)RVQ_L * RVQ_B);
}
